// PointTransformerEmbedding_1984274891516
// MI455X (gfx1250) — compile-verified, matrix operand order checked by reading, not run on hardware
//
#include <hip/hip_runtime.h>
#include <math.h>

// ---------------------------------------------------------------------------
// PointTransformerEmbedding forward, CDNA5 (gfx1250, wave32, WMMA).
//
// Input index map assumes jax pytree flattening of setup_inputs() (dict keys
// sorted lexicographically at every level, lists in order):
//   top:   "params" < "pos" < "x"
//   params:"blocks" < "mlp_in" < "out1" < "out2" < "td"
//   block: att1{b,w} att2{b,w} dst{w} lin_in{b,w} lin_out{b,w}
//          pos1{b,w} pos2{b,w} src{w} val{w}            (15 leaves / block)
//   => d_in[0..14] block0, [15..29] block1, [30..44] block2,
//      [45] mlp_in.b [46] mlp_in.w [47] out1.b [48] out1.w
//      [49] out2.b [50] out2.w [51] td0.b [52] td0.w [53] td1.b [54] td1.w
//      [55] pos (4*4096*2 f32), [56] x (4*4096*1 f32)
// Output: [4,64] f32.
// ---------------------------------------------------------------------------

typedef __attribute__((ext_vector_type(16))) _Float16 v16h;
typedef __attribute__((ext_vector_type(8)))  _Float16 v8h;
typedef __attribute__((ext_vector_type(8)))  float    v8f;

__device__ __forceinline__ int kfrag(int e, int half) {
  // K index of element e of a 16-bit WMMA A/B fragment for lane-half `half`
  return ((e >> 3) << 4) + (half << 3) + (e & 7);
}

// ---------------- weight f32 -> f16, packed in B-fragment order ------------
// Packed element gi = (((ntile*ksteps + kk)*32 + lane)*16 + e); each lane's 16
// halves are contiguous (32B) so a B fragment is one v16h load.
__global__ void conv_pack_kernel(const float* __restrict__ src,
                                 _Float16* __restrict__ dst,
                                 int Klogical, int N, int padK) {
  int gi = blockIdx.x * 256 + threadIdx.x;
  int ksteps = padK >> 5;
  int total = (N >> 4) * ksteps * 32 * 16;
  if (gi >= total) return;
  int e = gi & 15;
  int lane = (gi >> 4) & 31;
  int kk = (gi >> 9) % ksteps;
  int ntile = (gi >> 9) / ksteps;
  int n = (ntile << 4) + (lane & 15);
  int k = (kk << 5) + kfrag(e, lane >> 4);
  dst[gi] = (k < Klogical) ? (_Float16)src[(size_t)k * N + n] : (_Float16)0.0f;
}

// ------------------------------- mlp_in ------------------------------------
__global__ void mlp_in_kernel(const float* __restrict__ x,
                              const float* __restrict__ w,
                              const float* __restrict__ b,
                              float* __restrict__ h0, int nNodes) {
  int gi = blockIdx.x * 256 + threadIdx.x;
  if (gi >= nNodes * 32) return;
  int c = gi & 31;
  int i = gi >> 5;
  h0[gi] = fmaxf(x[i] * w[c] + b[c], 0.0f);
}

// --------------------------- generic WMMA GEMM -----------------------------
// Out[M,N] = act(A[M,K](f32) * Wp(packed f16) + bias). K%32==0, N%16==0.
// A row index clamped to M-1 for loads (garbage rows never stored).
template <bool RELU>
__global__ __launch_bounds__(128) void gemm_kernel(
    const float* __restrict__ A, const _Float16* __restrict__ Wp,
    const float* __restrict__ bias, float* __restrict__ Out,
    int M, int K, int N) {
  int nt = N >> 4;
  int mt = (M + 15) >> 4;
  int tile = blockIdx.x * 4 + (threadIdx.x >> 5);
  if (tile >= mt * nt) return;
  int mtile = tile / nt;
  int ntile = tile % nt;
  int lane = threadIdx.x & 31;
  int half = lane >> 4;
  int l16 = lane & 15;
  int m = (mtile << 4) + l16;
  int mld = min(m, M - 1);
  int n = (ntile << 4) + l16;
  int ksteps = K >> 5;
  const float* Ar = A + (size_t)mld * K + (half << 3);
  const _Float16* wp = Wp + ((size_t)ntile * ksteps * 32 + lane) * 16;
  v8f c = {0.f, 0.f, 0.f, 0.f, 0.f, 0.f, 0.f, 0.f};
  for (int kk = 0; kk < ksteps; ++kk) {
    const float4* ap0 = (const float4*)(Ar + (kk << 5));
    const float4* ap1 = (const float4*)(Ar + (kk << 5) + 16);
    float4 a0 = ap0[0], a1 = ap0[1];
    float4 a2 = ap1[0], a3 = ap1[1];
    v16h a;
    a[0] = (_Float16)a0.x;  a[1] = (_Float16)a0.y;
    a[2] = (_Float16)a0.z;  a[3] = (_Float16)a0.w;
    a[4] = (_Float16)a1.x;  a[5] = (_Float16)a1.y;
    a[6] = (_Float16)a1.z;  a[7] = (_Float16)a1.w;
    a[8] = (_Float16)a2.x;  a[9] = (_Float16)a2.y;
    a[10] = (_Float16)a2.z; a[11] = (_Float16)a2.w;
    a[12] = (_Float16)a3.x; a[13] = (_Float16)a3.y;
    a[14] = (_Float16)a3.z; a[15] = (_Float16)a3.w;
    v16h bf = *(const v16h*)(wp + (size_t)kk * 512);
    c = __builtin_amdgcn_wmma_f32_16x16x32_f16(false, a, false, bf, (short)0,
                                               c, false, false);
  }
#pragma unroll
  for (int r = 0; r < 8; ++r) {
    int mm = (mtile << 4) + (half << 3) + r;
    if (mm < M) {
      float v = c[r] + (bias ? bias[n] : 0.0f);
      if (RELU) v = fmaxf(v, 0.0f);
      Out[(size_t)mm * N + n] = v;
    }
  }
}

// ------------------------------- KNN ---------------------------------------
// For each query, KN smallest squared distances to ref (ties -> lower index,
// matching lax.top_k). EXCL skips self (q==ref). APPEND writes self at [KN].
template <int KN, bool EXCL, bool APPEND>
__global__ void knn_kernel(const float* __restrict__ q,
                           const float* __restrict__ ref,
                           int nq, int nr, int* __restrict__ out) {
  int perB = (nq + 255) / 256;
  int b = blockIdx.x / perB;
  int ti = (blockIdx.x % perB) * 256 + threadIdx.x;
  __shared__ float2 sref[256];
  float qx = 0.f, qy = 0.f;
  if (ti < nq) {
    qx = q[((size_t)b * nq + ti) * 2 + 0];
    qy = q[((size_t)b * nq + ti) * 2 + 1];
  }
  float best[KN];
  int bidx[KN];
#pragma unroll
  for (int s = 0; s < KN; ++s) { best[s] = 1e30f; bidx[s] = 0; }
  for (int base = 0; base < nr; base += 256) {
    __syncthreads();
    int j = base + threadIdx.x;
    if (j < nr) {
      sref[threadIdx.x] =
          make_float2(ref[((size_t)b * nr + j) * 2 + 0],
                      ref[((size_t)b * nr + j) * 2 + 1]);
    }
    __syncthreads();
    int lim = min(256, nr - base);
    for (int jj = 0; jj < lim; ++jj) {
      int idx = base + jj;
      if (EXCL && idx == ti) continue;
      float dx = qx - sref[jj].x, dy = qy - sref[jj].y;
      float d = dx * dx + dy * dy;
      if (d < best[KN - 1]) {
        best[KN - 1] = d;
        bidx[KN - 1] = idx;
#pragma unroll
        for (int s = KN - 1; s > 0; --s) {
          if (best[s] < best[s - 1]) {
            float tv = best[s]; best[s] = best[s - 1]; best[s - 1] = tv;
            int tii = bidx[s]; bidx[s] = bidx[s - 1]; bidx[s - 1] = tii;
          }
        }
      }
    }
  }
  if (ti < nq) {
    int* o = out + ((size_t)b * nq + ti) * (APPEND ? KN + 1 : KN);
#pragma unroll
    for (int s = 0; s < KN; ++s) o[s] = bidx[s];
    if (APPEND) o[KN] = ti;
  }
}

// ------------------------------- FPS ---------------------------------------
__global__ void fps_kernel(const float* __restrict__ pos, int Nn, int m,
                           int* __restrict__ idc) {
  int b = blockIdx.x;
  __shared__ float mind[4096];
  __shared__ float rv[256];
  __shared__ int ri[256];
  int tid = threadIdx.x;
  const float* P = pos + (size_t)b * Nn * 2;
  float x0 = P[0], y0 = P[1];
  for (int j = tid; j < Nn; j += 256) {
    float dx = P[j * 2] - x0, dy = P[j * 2 + 1] - y0;
    mind[j] = dx * dx + dy * dy;
  }
  if (tid == 0) idc[(size_t)b * m] = 0;
  __syncthreads();
  for (int s = 1; s < m; ++s) {
    float bv = -1e30f;
    int bi = 0;
    for (int j = tid; j < Nn; j += 256) {
      float v = mind[j];
      if (v > bv) { bv = v; bi = j; }   // ascending j + strict > => lowest idx
    }
    rv[tid] = bv; ri[tid] = bi;
    __syncthreads();
    for (int off = 128; off > 0; off >>= 1) {
      if (tid < off) {
        if (rv[tid + off] > rv[tid] ||
            (rv[tid + off] == rv[tid] && ri[tid + off] < ri[tid])) {
          rv[tid] = rv[tid + off];
          ri[tid] = ri[tid + off];
        }
      }
      __syncthreads();
    }
    int pi = ri[0];
    if (tid == 0) idc[(size_t)b * m + s] = pi;
    float px = P[pi * 2], py = P[pi * 2 + 1];
    for (int j = tid; j < Nn; j += 256) {
      float dx = P[j * 2] - px, dy = P[j * 2 + 1] - py;
      mind[j] = fminf(mind[j], dx * dx + dy * dy);
    }
    __syncthreads();
  }
}

__global__ void gather_pos_kernel(const float* __restrict__ pos,
                                  const int* __restrict__ idc,
                                  float* __restrict__ sub, int msub, int Nn) {
  int gi = blockIdx.x * 256 + threadIdx.x;
  if (gi >= 4 * msub) return;
  int b = gi / msub, i = gi % msub;
  int j = idc[b * msub + i];
  sub[((size_t)b * msub + i) * 2 + 0] = pos[((size_t)b * Nn + j) * 2 + 0];
  sub[((size_t)b * msub + i) * 2 + 1] = pos[((size_t)b * Nn + j) * 2 + 1];
}

// --------------------- InstanceNorm(+ReLU) per (b,c) -----------------------
__global__ void instnorm_relu_kernel(float* __restrict__ h, int Mn, int C) {
  int b = blockIdx.x / C, c = blockIdx.x % C;
  __shared__ float s1[256], s2[256];
  int tid = threadIdx.x;
  float a = 0.f, a2 = 0.f;
  for (int j = tid; j < Mn; j += 256) {
    float v = h[((size_t)b * Mn + j) * C + c];
    a += v; a2 += v * v;
  }
  s1[tid] = a; s2[tid] = a2;
  __syncthreads();
  for (int off = 128; off > 0; off >>= 1) {
    if (tid < off) { s1[tid] += s1[tid + off]; s2[tid] += s2[tid + off]; }
    __syncthreads();
  }
  float mu = s1[0] / (float)Mn;
  float var = s2[0] / (float)Mn - mu * mu;
  float inv = rsqrtf(var + 1e-5f);
  for (int j = tid; j < Mn; j += 256) {
    size_t o = ((size_t)b * Mn + j) * C + c;
    h[o] = fmaxf((h[o] - mu) * inv, 0.0f);
  }
}

__global__ void gathermax_kernel(const float* __restrict__ h,
                                 const int* __restrict__ nbr,
                                 float* __restrict__ out,
                                 int msub, int Mn, int C) {
  size_t gi = (size_t)blockIdx.x * 256 + threadIdx.x;
  if (gi >= (size_t)4 * msub * C) return;
  int c = (int)(gi % C);
  size_t t = gi / C;
  int i = (int)(t % msub);
  int b = (int)(t / msub);
  float mx = -1e30f;
  for (int j = 0; j < 17; ++j) {
    int jd = nbr[((size_t)b * msub + i) * 17 + j];
    mx = fmaxf(mx, h[((size_t)b * Mn + jd) * C + c]);
  }
  out[((size_t)b * msub + i) * C + c] = mx;
}

__global__ void meanpool_kernel(const float* __restrict__ h,
                                float* __restrict__ g, int Mn, int C) {
  int b = blockIdx.x;
  int c = threadIdx.x;
  if (c >= C) return;
  float s = 0.f;
  for (int j = 0; j < Mn; ++j) s += h[((size_t)b * Mn + j) * C + c];
  g[b * C + c] = s / (float)Mn;
}

// ------------------- fused per-edge attention (WMMA, LDS) ------------------
// LDS GEMM: Out = relu(A * W + bias).  A: LDS f16 row-major (lda, 16B-aligned
// rows), Wp: packed global f16 (B-fragment order), K = ksteps*32.
template <int MT>
__device__ __forceinline__ void lds_gemm(const _Float16* A, int lda,
                                         const _Float16* __restrict__ Wp,
                                         int ksteps, int ntOff, int N,
                                         const float* __restrict__ bias,
                                         _Float16* outH, float* outF, int ldo) {
  int wave = threadIdx.x >> 5;
  int nwaves = blockDim.x >> 5;
  int lane = threadIdx.x & 31;
  int half = lane >> 4;
  int l16 = lane & 15;
  int nt = N >> 4;
  for (int job = wave; job < MT * nt; job += nwaves) {
    int mtile = job / nt, ntile = job % nt;
    const _Float16* Ar = A + (mtile * 16 + l16) * lda + (half << 3);
    const _Float16* wp =
        Wp + ((size_t)(ntOff + ntile) * ksteps * 32 + lane) * 16;
    v8f c = {0.f, 0.f, 0.f, 0.f, 0.f, 0.f, 0.f, 0.f};
    for (int kk = 0; kk < ksteps; ++kk) {
      v8h lo = *(const v8h*)(Ar + (kk << 5));
      v8h hi = *(const v8h*)(Ar + (kk << 5) + 16);
      v16h a = __builtin_shufflevector(lo, hi, 0, 1, 2, 3, 4, 5, 6, 7, 8, 9,
                                       10, 11, 12, 13, 14, 15);
      v16h bf = *(const v16h*)(wp + (size_t)kk * 512);
      c = __builtin_amdgcn_wmma_f32_16x16x32_f16(false, a, false, bf, (short)0,
                                                 c, false, false);
    }
    int n = ntile * 16 + l16;
    float bb = bias[n];
#pragma unroll
    for (int r = 0; r < 8; ++r) {
      int mm = mtile * 16 + half * 8 + r;
      float v = fmaxf(c[r] + bb, 0.0f);
      if (outH) outH[mm * ldo + n] = (_Float16)v;
      else      outF[mm * ldo + n] = v;
    }
  }
}

// One workgroup handles 4 target nodes (68 edges, padded to 80 = 5 WMMA
// row-tiles). Whole pos-MLP -> attention-MLP -> softmax -> weighted-max chain
// stays in LDS; only the final [M,C] message hits HBM.
template <int C>
__global__ __launch_bounds__(128) void edge_kernel(
    const float* __restrict__ pos, const int* __restrict__ nbr,
    const float* __restrict__ asrc, const float* __restrict__ adst,
    const float* __restrict__ vv,
    const _Float16* __restrict__ pos1W, const float* __restrict__ pos1b,
    const _Float16* __restrict__ pos2W, const float* __restrict__ pos2b,
    const _Float16* __restrict__ att1W, const float* __restrict__ att1b,
    const _Float16* __restrict__ att2W, const float* __restrict__ att2b,
    float* __restrict__ outmsg, int Mn) {
  constexpr int ROWS = 80, RUSE = 68;
  __shared__ __align__(16) _Float16 sA[ROWS * C];   // delta, then alphaIn
  __shared__ __align__(16) _Float16 sPD[ROWS * C];  // pos-MLP output (f16)
  __shared__ __align__(16) _Float16 sH[ROWS * 64];  // pos1 / att1 hidden
  __shared__ __align__(16) float sAlpha[ROWS * 32]; // att2 out, 32-col chunk
  __shared__ int sIdx[ROWS];
  int tilesPer = Mn / 4;
  int b = blockIdx.x / tilesPer;
  int t0 = (blockIdx.x % tilesPer) * 4;
  int tid = threadIdx.x;

  for (int r = tid; r < ROWS; r += 128)
    sIdx[r] = (r < RUSE)
                  ? nbr[((size_t)b * Mn + t0 + r / 17) * 17 + (r % 17)]
                  : 0;
  __syncthreads();
  // delta = pos_i - pos_j, zero-padded to K=32 in sA (lda = C)
  for (int q = tid; q < ROWS * 32; q += 128) {
    int r = q >> 5, col = q & 31;
    float v = 0.f;
    if (r < RUSE && col < 2) {
      int i = t0 + r / 17;
      int j = sIdx[r];
      v = pos[((size_t)b * Mn + i) * 2 + col] -
          pos[((size_t)b * Mn + j) * 2 + col];
    }
    sA[r * C + col] = (_Float16)v;
  }
  __syncthreads();
  lds_gemm<5>(sA, C, pos1W, 1, 0, 64, pos1b, sH, nullptr, 64);     // 2->64
  __syncthreads();
  lds_gemm<5>(sH, 64, pos2W, 2, 0, C, pos2b, sPD, nullptr, C);     // 64->C
  __syncthreads();
  // alphaIn = a_dst[i] - a_src[j] + pdelta
  for (int q = tid; q < ROWS * C; q += 128) {
    int r = q / C, col = q % C;
    float v = 0.f;
    if (r < RUSE) {
      size_t i = (size_t)b * Mn + t0 + r / 17;
      size_t j = (size_t)b * Mn + sIdx[r];
      v = adst[i * C + col] - asrc[j * C + col] + (float)sPD[r * C + col];
    }
    sA[r * C + col] = (_Float16)v;
  }
  __syncthreads();
  lds_gemm<5>(sA, C, att1W, C / 32, 0, 64, att1b, sH, nullptr, 64); // C->64
  __syncthreads();
  for (int c0 = 0; c0 < C; c0 += 32) {
    lds_gemm<5>(sH, 64, att2W, 2, c0 >> 4, 32, att2b + c0,
                nullptr, sAlpha, 32);                               // 64->32
    __syncthreads();
    {
      int t = tid >> 5, cc = tid & 31;   // 4 targets x 32 channels = 128 thr
      int rb = t * 17;
      int cg = c0 + cc;
      float amax = -1e30f;
#pragma unroll
      for (int j = 0; j < 17; ++j)
        amax = fmaxf(amax, sAlpha[(rb + j) * 32 + cc]);
      float ee[17];
      float s = 0.f;
#pragma unroll
      for (int j = 0; j < 17; ++j) {
        float e = __expf(sAlpha[(rb + j) * 32 + cc] - amax);
        ee[j] = e;
        s += e;
      }
      float inv = 1.0f / s;
      float mx = -1e30f;
#pragma unroll
      for (int j = 0; j < 17; ++j) {
        size_t jg = (size_t)b * Mn + sIdx[rb + j];
        float val = ee[j] * inv *
                    (vv[jg * C + cg] + (float)sPD[(rb + j) * C + cg]);
        mx = fmaxf(mx, val);
      }
      outmsg[((size_t)b * Mn + t0 + t) * C + cg] = mx;
    }
    __syncthreads();
  }
}

// ---------------------------------------------------------------------------
extern "C" void kernel_launch(void* const* d_in, const int* in_sizes, int n_in,
                              void* d_out, int out_size, void* d_ws,
                              size_t ws_size, hipStream_t stream) {
  (void)in_sizes; (void)out_size;
  if (n_in < 57) return;
  const int B = 4, N0 = 4096, N1 = 1024, N2 = 256;
  const int Cs[3] = {32, 64, 128};
  const int Mns[3] = {N0, N1, N2};

  auto F = [&](int i) { return (const float*)d_in[i]; };
  const float* posIn = F(55);
  const float* xIn = F(56);

  // ---- workspace arena ----
  char* base = (char*)d_ws;
  size_t off = 0;
  auto bump = [&](size_t bytes) -> void* {
    void* p = base + off;
    off += (bytes + 255) & ~(size_t)255;
    return p;
  };
  float* h0 = (float*)bump((size_t)16384 * 32 * 4);
  float* G[6];
  for (int i = 0; i < 6; ++i) G[i] = (float*)bump((size_t)16384 * 128 * 4);
  int* nbrA = (int*)bump((size_t)16384 * 17 * 4);
  int* nbrB = (int*)bump((size_t)16384 * 17 * 4);
  int* idc = (int*)bump((size_t)4 * 1024 * 4);
  float* posL1 = (float*)bump((size_t)4 * 1024 * 2 * 4);
  float* posL2 = (float*)bump((size_t)4 * 256 * 2 * 4);
  float* gbuf = (float*)bump(4 * 128 * 4);
  float* ghb = (float*)bump(4 * 128 * 4);
  _Float16* w16base = (_Float16*)bump(512 * 1024);
  size_t w16off = 0;
  auto bump16 = [&](size_t elems) -> _Float16* {
    _Float16* p = w16base + w16off;
    w16off += (elems + 15) & ~(size_t)15;  // keep 32B alignment
    return p;
  };
  if (off > ws_size) return;  // insufficient scratch: do nothing (safe)

  // pack W[K,N] (f32, row-major) -> f16 B-fragment order, K padded to padK
  auto conv = [&](const float* src, _Float16* dst, int Klog, int N, int padK) {
    int n = padK * N;
    conv_pack_kernel<<<dim3((n + 255) / 256), 256, 0, stream>>>(src, dst, Klog,
                                                                N, padK);
  };

  struct BW {
    const float *att1b, *att2b, *linib, *linob, *pos1b, *pos2b;
    _Float16 *att1W, *att2W, *dstW, *liniW, *linoW, *pos1W, *pos2W, *srcW,
        *valW;
  } bw[3];
  for (int l = 0; l < 3; ++l) {
    int c = Cs[l];
    int ib = l * 15;
    bw[l].att1b = F(ib + 0);
    bw[l].att2b = F(ib + 2);
    bw[l].linib = F(ib + 5);
    bw[l].linob = F(ib + 7);
    bw[l].pos1b = F(ib + 9);
    bw[l].pos2b = F(ib + 11);
    bw[l].att1W = bump16((size_t)c * 64);
    bw[l].att2W = bump16((size_t)64 * c);
    bw[l].dstW = bump16((size_t)c * c);
    bw[l].liniW = bump16((size_t)c * c);
    bw[l].linoW = bump16((size_t)c * c);
    bw[l].pos1W = bump16((size_t)32 * 64);  // K padded 2 -> 32
    bw[l].pos2W = bump16((size_t)64 * c);
    bw[l].srcW = bump16((size_t)c * c);
    bw[l].valW = bump16((size_t)c * c);
    conv(F(ib + 1), bw[l].att1W, c, 64, c);
    conv(F(ib + 3), bw[l].att2W, 64, c, 64);
    conv(F(ib + 4), bw[l].dstW, c, c, c);
    conv(F(ib + 6), bw[l].liniW, c, c, c);
    conv(F(ib + 8), bw[l].linoW, c, c, c);
    conv(F(ib + 10), bw[l].pos1W, 2, 64, 32);
    conv(F(ib + 12), bw[l].pos2W, 64, c, 64);
    conv(F(ib + 13), bw[l].srcW, c, c, c);
    conv(F(ib + 14), bw[l].valW, c, c, c);
  }
  _Float16* td0W = bump16(32 * 64);
  _Float16* td1W = bump16(64 * 128);
  _Float16* out1W = bump16(128 * 128);
  _Float16* out2W = bump16(128 * 64);
  conv(F(52), td0W, 32, 64, 32);
  conv(F(54), td1W, 64, 128, 64);
  conv(F(48), out1W, 128, 128, 128);
  conv(F(50), out2W, 128, 64, 128);

  auto gemm = [&](bool relu, const float* A, const _Float16* W,
                  const float* bias, float* Out, int M, int K, int N) {
    int tiles = ((M + 15) / 16) * (N / 16);
    dim3 grid((tiles + 3) / 4);
    if (relu)
      gemm_kernel<true><<<grid, 128, 0, stream>>>(A, W, bias, Out, M, K, N);
    else
      gemm_kernel<false><<<grid, 128, 0, stream>>>(A, W, bias, Out, M, K, N);
  };

  auto run_block = [&](int lvl, const float* posP, const int* nbrP,
                       const float* hin, float* xact, float* asrc, float* adst,
                       float* vvb, float* msg, float* hout) {
    int C = Cs[lvl], Mn = Mns[lvl], M = B * Mn;
    const BW& w = bw[lvl];
    gemm(true, hin, w.liniW, w.linib, xact, M, C, C);
    gemm(false, xact, w.srcW, nullptr, asrc, M, C, C);
    gemm(false, xact, w.dstW, nullptr, adst, M, C, C);
    gemm(false, xact, w.valW, nullptr, vvb, M, C, C);
    dim3 eg(B * (Mn / 4));
    if (C == 32)
      edge_kernel<32><<<eg, 128, 0, stream>>>(
          posP, nbrP, asrc, adst, vvb, w.pos1W, w.pos1b, w.pos2W, w.pos2b,
          w.att1W, w.att1b, w.att2W, w.att2b, msg, Mn);
    else if (C == 64)
      edge_kernel<64><<<eg, 128, 0, stream>>>(
          posP, nbrP, asrc, adst, vvb, w.pos1W, w.pos1b, w.pos2W, w.pos2b,
          w.att1W, w.att1b, w.att2W, w.att2b, msg, Mn);
    else
      edge_kernel<128><<<eg, 128, 0, stream>>>(
          posP, nbrP, asrc, adst, vvb, w.pos1W, w.pos1b, w.pos2W, w.pos2b,
          w.att1W, w.att1b, w.att2W, w.att2b, msg, Mn);
    gemm(true, msg, w.linoW, w.linob, hout, M, C, C);
  };

  // ---- level 0 ----
  mlp_in_kernel<<<dim3((16384 * 32 + 255) / 256), 256, 0, stream>>>(
      xIn, F(46), F(45), h0, 16384);
  knn_kernel<16, true, true><<<dim3(B * (N0 / 256)), 256, 0, stream>>>(
      posIn, posIn, N0, N0, nbrA);
  run_block(0, posIn, nbrA, h0, G[0], G[1], G[2], G[3], G[4], G[5]);

  // ---- transition down 0 ----
  fps_kernel<<<dim3(B), 256, 0, stream>>>(posIn, N0, N1, idc);
  gather_pos_kernel<<<dim3((4 * N1 + 255) / 256), 256, 0, stream>>>(
      posIn, idc, posL1, N1, N0);
  knn_kernel<17, false, false><<<dim3(B * (N1 / 256)), 256, 0, stream>>>(
      posL1, posIn, N1, N0, nbrB);
  gemm(false, G[5], td0W, F(51), G[0], B * N0, 32, 64);
  instnorm_relu_kernel<<<dim3(B * 64), 256, 0, stream>>>(G[0], N0, 64);
  gathermax_kernel<<<dim3((4 * N1 * 64 + 255) / 256), 256, 0, stream>>>(
      G[0], nbrB, G[1], N1, N0, 64);

  // ---- level 1 ----
  knn_kernel<16, true, true><<<dim3(B * (N1 / 256)), 256, 0, stream>>>(
      posL1, posL1, N1, N1, nbrA);
  run_block(1, posL1, nbrA, G[1], G[0], G[2], G[3], G[4], G[5], G[0]);

  // ---- transition down 1 ----
  fps_kernel<<<dim3(B), 256, 0, stream>>>(posL1, N1, N2, idc);
  gather_pos_kernel<<<dim3((4 * N2 + 255) / 256), 256, 0, stream>>>(
      posL1, idc, posL2, N2, N1);
  knn_kernel<17, false, false><<<dim3(B * 1), 256, 0, stream>>>(
      posL2, posL1, N2, N1, nbrB);
  gemm(false, G[0], td1W, F(53), G[1], B * N1, 64, 128);
  instnorm_relu_kernel<<<dim3(B * 128), 256, 0, stream>>>(G[1], N1, 128);
  gathermax_kernel<<<dim3((4 * N2 * 128 + 255) / 256), 256, 0, stream>>>(
      G[1], nbrB, G[2], N2, N1, 128);

  // ---- level 2 ----
  knn_kernel<16, true, true><<<dim3(B * 1), 256, 0, stream>>>(
      posL2, posL2, N2, N2, nbrA);
  run_block(2, posL2, nbrA, G[2], G[0], G[1], G[3], G[4], G[5], G[0]);

  // ---- head ----
  meanpool_kernel<<<dim3(B), 128, 0, stream>>>(G[0], gbuf, N2, 128);
  gemm(true, gbuf, out1W, F(47), ghb, 4, 128, 128);
  gemm(false, ghb, out2W, F(49), (float*)d_out, 4, 128, 64);
}
